// GQAttention_66151086293119
// MI455X (gfx1250) — compile-verified, matrix operand order checked by reading, not run on hardware
//
#include <hip/hip_runtime.h>
#include <stdint.h>

// ---------------- problem constants ----------------
constexpr int BB  = 2;
constexpr int NN  = 2048;
constexpr int CC  = 2048;
constexpr int HH  = 16;
constexpr int KVH = 4;
constexpr int DD  = 128;           // head dim
constexpr int KVC = KVH * DD;      // 512

typedef __attribute__((ext_vector_type(16))) __bf16 bf16x16;
typedef __attribute__((ext_vector_type(8)))  float  f32x8;
typedef __attribute__((ext_vector_type(4)))  unsigned int uint32x4;
typedef __attribute__((ext_vector_type(8)))  unsigned int uint32x8;

union FragU {
    unsigned int u[8];
    bf16x16      b;
};

__device__ __forceinline__ unsigned short f32_to_bf16(float f) {
    unsigned int u = __builtin_bit_cast(unsigned int, f);
    u += 0x7FFFu + ((u >> 16) & 1u);   // round-to-nearest-even
    return (unsigned short)(u >> 16);
}

// K-pair offset pattern for 16-bit WMMA A/B fragments (per ISA 7.12.2):
// vgpr v (0..7), half hi (0/1): pair base k = (v>>2)*16 + hi*8 + (v&3)*2
__device__ __forceinline__ int kpat(int v, int hi) {
    return ((v >> 2) << 4) + (hi << 3) + ((v & 3) << 1);
}

// ---------------- elementwise conversion kernels ----------------
__global__ void cvt_f32_bf16_kernel(const float* __restrict__ in,
                                    unsigned short* __restrict__ out, int total) {
    int i = blockIdx.x * blockDim.x + threadIdx.x;
    if (i < total) out[i] = f32_to_bf16(in[i]);
}

// W (K x Nc, row-major f32)  ->  Wt (Nc x K, row-major bf16)
__global__ void transpose_cvt_kernel(const float* __restrict__ W,
                                     unsigned short* __restrict__ Wt,
                                     int K, int Nc) {
    int i = blockIdx.x * blockDim.x + threadIdx.x;
    if (i >= K * Nc) return;
    int k = i / Nc, n = i % Nc;
    Wt[(size_t)n * K + k] = f32_to_bf16(W[i]);
}

// ---------------- bf16 WMMA GEMM with TDM-staged weight tiles ----------------
// Out[M,Nc] (f32) = A[M,K](bf16) * Wt[Nc,K](bf16)^T + bias
// Block = 4 waves = 64x64 output tile (wave w owns rows m0+16w..m0+16w+15).
// The shared 64(n) x 32(k) weight tile lives in LDS; each wave TDM-loads its
// own 16-row slice, double-buffered across k-chunks (TENSORcnt + barrier).
__global__ void gemm_bf16_tdm_kernel(const unsigned short* __restrict__ A,
                                     const unsigned short* __restrict__ Wt,
                                     const float* __restrict__ bias,
                                     float* __restrict__ Out,
                                     int M, int K, int Nc) {
    __shared__ unsigned short ldsB[2][64 * 32];

    const int wid  = threadIdx.x >> 5;      // wave 0..3
    const int lane = threadIdx.x & 31;
    const int lm   = lane & 15;
    const int lh   = lane >> 4;

    const int nBlocks = Nc >> 6;
    const int m0 = (blockIdx.x / nBlocks) * 64 + wid * 16;   // this wave's rows
    const int n0 = (blockIdx.x % nBlocks) * 64;              // block's columns

    // ---- TDM descriptor (D#) pieces -----------------------------------
    // group1 (constant): data_size=2B, tensor_dim0=K, tensor_dim1=16 rows,
    // tile_dim0=32, tile_dim1=16, tensor_dim0_stride=K.
    uint32x8 g1;
    g1[0] = 1u << 16;                                   // data_size = 2 bytes
    g1[1] = ((unsigned)K & 0xFFFFu) << 16;              // tensor_dim0 [15:0]
    g1[2] = (((unsigned)K >> 16) & 0xFFFFu)             // tensor_dim0 [31:16]
          | (16u << 16);                                // tensor_dim1 = 16 rows
    g1[3] = (32u << 16);                                // tile_dim0 = 32
    g1[4] = 16u;                                        // tile_dim1 = 16
    g1[5] = (unsigned)K;                                // dim0_stride [31:0]
    g1[6] = 0u;
    g1[7] = 0u;

    // wave's 16-row slab of the weight matrix
    const uint64_t wbase =
        (uint64_t)(uintptr_t)(Wt + (size_t)(n0 + wid * 16) * K);
    // LDS byte offset of this wave's quadrant (flat addr low 32 bits == offset)
    const unsigned int ldsq =
        (unsigned int)(uintptr_t)(&ldsB[0][0]) + (unsigned)(wid * 16 * 32 * 2);

    auto issue_tdm = [&](int kb, int buf) {
        uint64_t ga = wbase + (uint64_t)kb * 2u;
        uint32x4 g0;
        g0[0] = 1u;                                     // count=1 (valid D#)
        g0[1] = ldsq + (unsigned)buf * (64u * 32u * 2u);
        g0[2] = (unsigned)ga;
        g0[3] = (unsigned)((ga >> 32) & 0x01FFFFFFu) | (2u << 30);  // type=2
        asm volatile("tensor_load_to_lds %0, %1" : : "s"(g0), "s"(g1) : "memory");
    };

    f32x8 acc0 = {}, acc1 = {}, acc2 = {}, acc3 = {};
    const size_t arow = (size_t)(m0 + lm) * K;

    issue_tdm(0, 0);
    int buf = 0;
    for (int kb = 0; kb < K; kb += 32) {
        __builtin_amdgcn_s_wait_tensorcnt(0);   // my slice of current tile done
        __syncthreads();                        // whole 64x32 tile published
        if (kb + 32 < K) issue_tdm(kb + 32, buf ^ 1);   // prefetch next chunk

        FragU a;
#pragma unroll
        for (int v = 0; v < 8; ++v)
            a.u[v] = *(const unsigned int*)(A + arow + kb + kpat(v, lh));

        const unsigned short* Bb = &ldsB[buf][0];
#pragma unroll
        for (int t = 0; t < 4; ++t) {
            FragU bf;
#pragma unroll
            for (int v = 0; v < 8; ++v)
                bf.u[v] = *(const unsigned int*)(Bb + (t * 16 + lm) * 32 +
                                                 kpat(v, lh));
            f32x8 c = (t == 0) ? acc0 : (t == 1) ? acc1 : (t == 2) ? acc2 : acc3;
            c = __builtin_amdgcn_wmma_f32_16x16x32_bf16(false, a.b, false, bf.b,
                                                        (short)0, c, false, false);
            if (t == 0) acc0 = c; else if (t == 1) acc1 = c;
            else if (t == 2) acc2 = c; else acc3 = c;
        }
        buf ^= 1;
    }

#pragma unroll
    for (int t = 0; t < 4; ++t) {
        int n = n0 + t * 16 + lm;
        float bv = bias ? bias[n] : 0.0f;
        f32x8 c = (t == 0) ? acc0 : (t == 1) ? acc1 : (t == 2) ? acc2 : acc3;
#pragma unroll
        for (int r = 0; r < 8; ++r) {
            int m = m0 + r + lh * 8;
            Out[(size_t)m * Nc + n] = c[r] + bv;
        }
    }
}

// ---------------- RoPE + layout kernels ----------------
__global__ void rope_q_kernel(const float* __restrict__ Qf,
                              const int* __restrict__ pos_ids,
                              unsigned short* __restrict__ Qb, float qscale) {
    int i = blockIdx.x * blockDim.x + threadIdx.x;
    const int PD = DD / 2;
    if (i >= BB * NN * HH * PD) return;
    int d2 = i % PD;  i /= PD;
    int h  = i % HH;  i /= HH;
    int n  = i % NN;  int b = i / NN;

    float pos = (float)pos_ids[b * NN + n];
    float ang = pos * __powf(10000.0f, -(float)(2 * d2) / (float)DD);
    float s, c;
    __sincosf(ang, &s, &c);

    const float* src = Qf + ((size_t)(b * NN + n)) * CC + h * DD + 2 * d2;
    float x0 = src[0], x1 = src[1];
    float y0 = (x0 * c - x1 * s) * qscale;
    float y1 = (x1 * c + x0 * s) * qscale;

    unsigned short* dst = Qb + (((size_t)(b * HH + h)) * NN + n) * DD + 2 * d2;
    dst[0] = f32_to_bf16(y0);
    dst[1] = f32_to_bf16(y1);
}

__global__ void rope_k_kernel(const float* __restrict__ Kf,
                              const int* __restrict__ pos_ids,
                              unsigned short* __restrict__ Kb) {
    int i = blockIdx.x * blockDim.x + threadIdx.x;
    const int PD = DD / 2;
    if (i >= BB * NN * KVH * PD) return;
    int d2 = i % PD;   i /= PD;
    int kv = i % KVH;  i /= KVH;
    int n  = i % NN;   int b = i / NN;

    float pos = (float)pos_ids[b * NN + n];
    float ang = pos * __powf(10000.0f, -(float)(2 * d2) / (float)DD);
    float s, c;
    __sincosf(ang, &s, &c);

    const float* src = Kf + ((size_t)(b * NN + n)) * KVC + kv * DD + 2 * d2;
    float x0 = src[0], x1 = src[1];

    unsigned short* dst = Kb + (((size_t)(b * KVH + kv)) * NN + n) * DD + 2 * d2;
    dst[0] = f32_to_bf16(x0 * c - x1 * s);
    dst[1] = f32_to_bf16(x1 * c + x0 * s);
}

// Vf: (b, n, kv*D) f32 -> Vtb: (b, kv, D, N) bf16
__global__ void vt_cvt_kernel(const float* __restrict__ Vf,
                              unsigned short* __restrict__ Vtb) {
    int i = blockIdx.x * blockDim.x + threadIdx.x;
    if (i >= BB * NN * KVH * DD) return;
    int d  = i % DD;   i /= DD;
    int kv = i % KVH;  i /= KVH;
    int n  = i % NN;   int b = i / NN;
    float v = Vf[((size_t)(b * NN + n)) * KVC + kv * DD + d];
    Vtb[(((size_t)(b * KVH + kv)) * DD + d) * NN + n] = f32_to_bf16(v);
}

// ---------------- flash attention (causal, GQA) ----------------
__global__ void attn_kernel(const unsigned short* __restrict__ Qb,
                            const unsigned short* __restrict__ Kb,
                            const unsigned short* __restrict__ Vtb,
                            unsigned short* __restrict__ Ob) {
    __shared__ unsigned short ldsP[16 * 32];

    int bid = blockIdx.x;
    int qt  = bid % (NN / 16); bid /= (NN / 16);
    int h   = bid % HH;        int b = bid / HH;
    int kvh = h / (HH / KVH);
    int q0  = qt * 16;

    int lane = threadIdx.x & 31;
    int lm   = lane & 15;
    int lh   = lane >> 4;

    const unsigned short* Qbase = Qb + (((size_t)(b * HH + h)) * NN + q0) * DD;
    const unsigned short* Kbase = Kb + ((size_t)(b * KVH + kvh)) * NN * DD;
    const unsigned short* Vbase = Vtb + ((size_t)(b * KVH + kvh)) * DD * NN;

    FragU qf[4];
#pragma unroll
    for (int kc = 0; kc < 4; ++kc)
#pragma unroll
        for (int v = 0; v < 8; ++v)
            qf[kc].u[v] = *(const unsigned int*)(Qbase + (size_t)lm * DD +
                                                 kc * 32 + kpat(v, lh));

    f32x8 acc[8];
    float mi[8], lsum[8];
#pragma unroll
    for (int r = 0; r < 8; ++r) { mi[r] = -1e30f; lsum[r] = 0.0f; }
#pragma unroll
    for (int dt = 0; dt < 8; ++dt) acc[dt] = (f32x8){};

    const int jend = q0 + 16;
    for (int j = 0; j < jend; j += 32) {
        f32x8 st[2];
#pragma unroll
        for (int t = 0; t < 2; ++t) {
            f32x8 s = {};
            const size_t krow = (size_t)(j + t * 16 + lm) * DD;
#pragma unroll
            for (int kc = 0; kc < 4; ++kc) {
                FragU kf;
#pragma unroll
                for (int v = 0; v < 8; ++v)
                    kf.u[v] = *(const unsigned int*)(Kbase + krow + kc * 32 +
                                                     kpat(v, lh));
                s = __builtin_amdgcn_wmma_f32_16x16x32_bf16(false, qf[kc].b,
                        false, kf.b, (short)0, s, false, false);
            }
            st[t] = s;
        }

#pragma unroll
        for (int t = 0; t < 2; ++t) {
            int kvi = j + t * 16 + lm;
#pragma unroll
            for (int r = 0; r < 8; ++r) {
                int row = q0 + r + lh * 8;
                if (kvi > row) st[t][r] = -1e30f;
            }
        }

        float alpha[8];
#pragma unroll
        for (int r = 0; r < 8; ++r) {
            float v = fmaxf(st[0][r], st[1][r]);
#pragma unroll
            for (int off = 8; off >= 1; off >>= 1)
                v = fmaxf(v, __shfl_xor(v, off, 32));
            float mnew = fmaxf(mi[r], v);
            alpha[r] = __expf(mi[r] - mnew);
            mi[r] = mnew;
        }
#pragma unroll
        for (int t = 0; t < 2; ++t)
#pragma unroll
            for (int r = 0; r < 8; ++r)
                st[t][r] = __expf(st[t][r] - mi[r]);
#pragma unroll
        for (int r = 0; r < 8; ++r) {
            float ps = st[0][r] + st[1][r];
#pragma unroll
            for (int off = 8; off >= 1; off >>= 1)
                ps += __shfl_xor(ps, off, 32);
            lsum[r] = lsum[r] * alpha[r] + ps;
        }
#pragma unroll
        for (int dt = 0; dt < 8; ++dt)
#pragma unroll
            for (int r = 0; r < 8; ++r)
                acc[dt][r] *= alpha[r];

#pragma unroll
        for (int t = 0; t < 2; ++t)
#pragma unroll
            for (int r = 0; r < 8; ++r)
                ldsP[(r + lh * 8) * 32 + t * 16 + lm] = f32_to_bf16(st[t][r]);
        __syncthreads();

        FragU pf;
#pragma unroll
        for (int v = 0; v < 8; ++v)
            pf.u[v] = *(const unsigned int*)(&ldsP[lm * 32 + kpat(v, lh)]);
        __syncthreads();

#pragma unroll
        for (int dt = 0; dt < 8; ++dt) {
            FragU vf;
            const size_t vrow = (size_t)(dt * 16 + lm) * NN + j;
#pragma unroll
            for (int v = 0; v < 8; ++v)
                vf.u[v] = *(const unsigned int*)(Vbase + vrow + kpat(v, lh));
            acc[dt] = __builtin_amdgcn_wmma_f32_16x16x32_bf16(false, pf.b,
                          false, vf.b, (short)0, acc[dt], false, false);
        }
    }

    float inv[8];
#pragma unroll
    for (int r = 0; r < 8; ++r) inv[r] = 1.0f / lsum[r];
#pragma unroll
    for (int dt = 0; dt < 8; ++dt) {
        int d = dt * 16 + lm;
#pragma unroll
        for (int r = 0; r < 8; ++r) {
            int row = q0 + r + lh * 8;
            Ob[((size_t)b * NN + row) * CC + h * DD + d] =
                f32_to_bf16(acc[dt][r] * inv[r]);
        }
    }
}

// ---------------- host launch ----------------
extern "C" void kernel_launch(void* const* d_in, const int* in_sizes, int n_in,
                              void* d_out, int out_size, void* d_ws, size_t ws_size,
                              hipStream_t stream) {
    (void)in_sizes; (void)n_in; (void)out_size; (void)ws_size;
    const float* x    = (const float*)d_in[0];
    const int*   pid  = (const int*)d_in[1];
    const float* Wq   = (const float*)d_in[2];
    const float* bq   = (const float*)d_in[3];
    const float* Wk   = (const float*)d_in[4];
    const float* bk   = (const float*)d_in[5];
    const float* Wv   = (const float*)d_in[6];
    const float* bv   = (const float*)d_in[7];
    const float* Wo   = (const float*)d_in[8];
    float* out = (float*)d_out;

    size_t off = 0;
    auto alloc = [&](size_t bytes) {
        void* p = (char*)d_ws + off;
        off += (bytes + 255) & ~(size_t)255;
        return p;
    };
    const size_t M = (size_t)BB * NN;                  // 4096 rows
    unsigned short* xb  = (unsigned short*)alloc(M * CC * 2);
    unsigned short* WqT = (unsigned short*)alloc((size_t)CC * CC * 2);
    unsigned short* WkT = (unsigned short*)alloc((size_t)CC * KVC * 2);
    unsigned short* WvT = (unsigned short*)alloc((size_t)CC * KVC * 2);
    unsigned short* WoT = (unsigned short*)alloc((size_t)CC * CC * 2);
    float*          Qf  = (float*)alloc(M * CC * 4);
    float*          Kf  = (float*)alloc(M * KVC * 4);
    float*          Vf  = (float*)alloc(M * KVC * 4);
    unsigned short* Qb  = (unsigned short*)alloc(M * CC * 2);
    unsigned short* Kb  = (unsigned short*)alloc(M * KVC * 2);
    unsigned short* Vtb = (unsigned short*)alloc(M * KVC * 2);
    unsigned short* Ab  = (unsigned short*)alloc(M * CC * 2);

    {
        int total = (int)(M * CC);
        cvt_f32_bf16_kernel<<<total / 256, 256, 0, stream>>>(x, xb, total);
    }
    transpose_cvt_kernel<<<(CC * CC) / 256, 256, 0, stream>>>(Wq, WqT, CC, CC);
    transpose_cvt_kernel<<<(CC * KVC) / 256, 256, 0, stream>>>(Wk, WkT, CC, KVC);
    transpose_cvt_kernel<<<(CC * KVC) / 256, 256, 0, stream>>>(Wv, WvT, CC, KVC);
    transpose_cvt_kernel<<<(CC * CC) / 256, 256, 0, stream>>>(Wo, WoT, CC, CC);

    // QKV projections (WMMA + TDM): block = 64x64 tile, 128 threads
    gemm_bf16_tdm_kernel<<<((int)M / 64) * (CC / 64), 128, 0, stream>>>(
        xb, WqT, bq, Qf, (int)M, CC, CC);
    gemm_bf16_tdm_kernel<<<((int)M / 64) * (KVC / 64), 128, 0, stream>>>(
        xb, WkT, bk, Kf, (int)M, CC, KVC);
    gemm_bf16_tdm_kernel<<<((int)M / 64) * (KVC / 64), 128, 0, stream>>>(
        xb, WvT, bv, Vf, (int)M, CC, KVC);

    {
        int total = BB * NN * HH * (DD / 2);
        rope_q_kernel<<<total / 256, 256, 0, stream>>>(Qf, pid, Qb,
                                                       0.08838834764831845f);
    }
    {
        int total = BB * NN * KVH * (DD / 2);
        rope_k_kernel<<<total / 256, 256, 0, stream>>>(Kf, pid, Kb);
    }
    {
        int total = BB * NN * KVH * DD;
        vt_cvt_kernel<<<total / 256, 256, 0, stream>>>(Vf, Vtb);
    }

    attn_kernel<<<BB * HH * (NN / 16), 32, 0, stream>>>(Qb, Kb, Vtb, Ab);

    gemm_bf16_tdm_kernel<<<((int)M / 64) * (CC / 64), 128, 0, stream>>>(
        Ab, WoT, nullptr, out, (int)M, CC, CC);
}
